// CaptionModel_19456201851385
// MI455X (gfx1250) — compile-verified, matrix operand order checked by reading, not run on hardware
//
#include <hip/hip_runtime.h>
#include <math.h>

// ---------------------------------------------------------------------------
// CaptionModel GRU decoder on MI455X (gfx1250, wave32, WMMA).
// B=512, FEAT=2048, H=512, V=100, T=200 steps.
// bf16 WMMA 16x16x32, f32 accumulators, f32 recurrent state.
// - Hidden state kept both as f32 and as bf16 pre-packed in the wave32
//   A-fragment layout -> recurrence inner loop is pure 32B loads + WMMA.
// - Projection of h_t is fused into step t+1's kernel (extra blocks),
//   halving the sequential launch count: 200 step launches + 1 tail proj.
// ---------------------------------------------------------------------------

typedef __attribute__((ext_vector_type(16))) __bf16 v16bf;
typedef __attribute__((ext_vector_type(8)))  float  v8f;

static constexpr int Bn = 512;   // batch
static constexpr int Hn = 512;   // hidden
static constexpr int Fn = 2048;  // feat
static constexpr int Vn = 100;   // vocab
static constexpr int Tn = 200;   // steps
static constexpr int KT = Hn / 32;  // 16 k-tiles for K=512 GEMMs

// round-to-nearest-even f32 -> bf16 (integer ops only)
__device__ __forceinline__ __bf16 f2bf(float f) {
    union { float f; unsigned u; } v; v.f = f;
    unsigned r = v.u + 0x7FFFu + ((v.u >> 16) & 1u);
    unsigned short h = (unsigned short)(r >> 16);
    return __builtin_bit_cast(__bf16, h);
}

__device__ __forceinline__ float sigmoidf_(float x) {
    return 1.0f / (1.0f + __expf(-x));
}

// A-fragment (16x32 bf16, wave32): lane<16 -> row M=lane, K = 0..7 & 16..23;
// lane>=16 -> row M=lane-16, K = 8..15 & 24..31.
// Packed layout: [mtile][kt][lane][16 bf16]  (32 bytes contiguous per lane)
__device__ __forceinline__ v16bf load_a_packed(const __bf16* __restrict__ pk,
                                               int mtile, int kt, int lane) {
    return *(const v16bf*)(pk + (((size_t)mtile * KT + kt) * 32 + lane) * 16);
}

// On-the-fly A fragment from row-major fp32 (one-shot h0 GEMM only)
__device__ __forceinline__ v16bf load_a_f32(const float* __restrict__ base,
                                            int row0, int ld, int kt, int lane) {
    const int r  = row0 + (lane & 15);
    const int kb = kt * 32 + ((lane & 16) ? 8 : 0);
    const float4* p = (const float4*)(base + (size_t)r * ld + kb);
    float4 f0 = p[0], f1 = p[1], f2 = p[4], f3 = p[5];
    v16bf a;
    a[0] = f2bf(f0.x); a[1] = f2bf(f0.y); a[2]  = f2bf(f0.z); a[3]  = f2bf(f0.w);
    a[4] = f2bf(f1.x); a[5] = f2bf(f1.y); a[6]  = f2bf(f1.z); a[7]  = f2bf(f1.w);
    a[8] = f2bf(f2.x); a[9] = f2bf(f2.y); a[10] = f2bf(f2.z); a[11] = f2bf(f2.w);
    a[12]= f2bf(f3.x); a[13]= f2bf(f3.y); a[14] = f2bf(f3.z); a[15] = f2bf(f3.w);
    return a;
}

__device__ __forceinline__ int pack_k(int kt, int lane, int e) {
    const int khalf = (lane & 16) ? 8 : 0;
    return kt * 32 + khalf + ((e < 8) ? e : e + 8);
}

// ---------------- weight / input packing kernels (once per launch) ---------

// w_hp [FEAT, Hn] (B[k,n] = w_hp[k*Hn+n]) -> [nt(32)][kt(64)][lane][e]
__global__ void pack_whp_kernel(const float* __restrict__ w_hp, __bf16* __restrict__ dst) {
    int idx = blockIdx.x * blockDim.x + threadIdx.x;
    if (idx >= 32 * 64 * 512) return;
    int e = idx & 15, lane = (idx >> 4) & 31, kt = (idx >> 9) & 63, nt = idx >> 15;
    int n = nt * 16 + (lane & 15);
    int k = pack_k(kt, lane, e);
    dst[idx] = f2bf(w_hp[(size_t)k * Hn + n]);
}

// gates g=0..2 from w_ih [3H,H], g=3..5 from w_hh; B_g[k,n] = w[(gl*H+n)*H+k]
// packed [g(6)][nt(32)][kt(16)][lane][e]
__global__ void pack_wc_kernel(const float* __restrict__ w_ih,
                               const float* __restrict__ w_hh,
                               __bf16* __restrict__ dst) {
    int idx = blockIdx.x * blockDim.x + threadIdx.x;
    if (idx >= 6 * 32 * 16 * 512) return;
    int e = idx & 15, lane = (idx >> 4) & 31, kt = (idx >> 9) & 15;
    int nt = (idx >> 13) & 31, g = idx >> 18;
    int n = nt * 16 + (lane & 15);
    int k = pack_k(kt, lane, e);
    float v;
    if (g < 3) v = w_ih[((size_t)(g * Hn + n)) * Hn + k];
    else       v = w_hh[((size_t)((g - 3) * Hn + n)) * Hn + k];
    dst[idx] = f2bf(v);
}

// w_proj [H, V], zero-padded to 8x16 cols: packed [nt(8)][kt(16)][lane][e]
__global__ void pack_wproj_kernel(const float* __restrict__ w_proj, __bf16* __restrict__ dst) {
    int idx = blockIdx.x * blockDim.x + threadIdx.x;
    if (idx >= 8 * 16 * 512) return;
    int e = idx & 15, lane = (idx >> 4) & 31, kt = (idx >> 9) & 15, nt = idx >> 13;
    int n = nt * 16 + (lane & 15);
    int k = pack_k(kt, lane, e);
    dst[idx] = (n < Vn) ? f2bf(w_proj[(size_t)k * Vn + n]) : f2bf(0.0f);
}

// x0 = embed[SOS=0] broadcast over batch, directly in packed A layout
__global__ void pack_x0_kernel(const float* __restrict__ embed, __bf16* __restrict__ dst) {
    int idx = blockIdx.x * blockDim.x + threadIdx.x;
    if (idx >= 32 * 16 * 512) return;
    int e = idx & 15, lane = (idx >> 4) & 31, kt = (idx >> 9) & 15;
    int k = pack_k(kt, lane, e);     // batch row irrelevant: broadcast
    dst[idx] = f2bf(embed[k]);
}

// --- LDS transpose-pack: C/D tile layout -> packed A-fragment layout -------
__device__ __forceinline__ void tile_pack_store(__bf16* __restrict__ lds,
                                                __bf16* __restrict__ hpack,
                                                int tid, int bx, int by) {
    if (tid < 256) {
        const int tl  = tid & 31;
        const int fi  = tid >> 5;      // 0..7
        const int mtl = fi & 3;        // local mtile
        const int ktl = fi >> 2;       // local ktile (0..1)
        const int rloc = mtl * 16 + (tl & 15);
        const int kbl  = ktl * 32 + ((tl & 16) ? 8 : 0);
        v16bf out;
#pragma unroll
        for (int e = 0; e < 8; ++e) out[e]     = lds[rloc * 66 + kbl + e];
#pragma unroll
        for (int e = 0; e < 8; ++e) out[e + 8] = lds[rloc * 66 + kbl + 16 + e];
        const int mtg = bx * 4 + mtl;
        const int ktg = by * 2 + ktl;
        *(v16bf*)(hpack + (((size_t)mtg * KT + ktg) * 32 + tl) * 16) = out;
    }
}

// ---------------- h0 = feat @ w_hp + b_hp (WMMA), emits f32 + packed bf16 --

__global__ __launch_bounds__(512, 1) void h0_kernel(const float* __restrict__ feat,
                                                    const __bf16* __restrict__ whp,
                                                    const float* __restrict__ b_hp,
                                                    float* __restrict__ h0,
                                                    __bf16* __restrict__ h0pack) {
    __shared__ __bf16 lds[64 * 66];
    const int lane = threadIdx.x & 31;
    const int wave = threadIdx.x >> 5;
    const int mi = wave & 3, ni = wave >> 2;
    const int rowBase = blockIdx.x * 64 + mi * 16;
    const int ntg = blockIdx.y * 4 + ni;           // 0..31
    v8f acc = {};
#pragma unroll 2
    for (int kt = 0; kt < Fn / 32; ++kt) {
        v16bf a = load_a_f32(feat, rowBase, Fn, kt, lane);
        v16bf b = *(const v16bf*)(whp + (((size_t)ntg * 64 + kt) * 512) + lane * 16);
        acc = __builtin_amdgcn_wmma_f32_16x16x32_bf16(false, a, false, b, (short)0, acc, false, false);
    }
    const int n = ntg * 16 + (lane & 15);
    const float bias = b_hp[n];
    const int rowOff = rowBase + ((lane & 16) ? 8 : 0);
    const int rl0 = mi * 16 + ((lane & 16) ? 8 : 0);
    const int cl  = ni * 16 + (lane & 15);
#pragma unroll
    for (int e = 0; e < 8; ++e) {
        const float v = acc[e] + bias;
        h0[(size_t)(rowOff + e) * Hn + n] = v;
        lds[(rl0 + e) * 66 + cl] = f2bf(v);
    }
    __syncthreads();
    tile_pack_store(lds, h0pack, threadIdx.x, blockIdx.x, blockIdx.y);
}

// ---------------- fused GRU step + projection of the INPUT hidden ----------
// grid (8,9), block 512 (16 waves).
//   blockIdx.y < 8 : gate GEMMs + GRU nonlinearity -> h_{t+1} (f32 + packed)
//   blockIdx.y == 8: project h_t (the packed input) -> out[:, :, t-1]
//                    (skipped when t==0; h_0 is not an output)

__global__ __launch_bounds__(512, 1) void gru_step_kernel(const __bf16* __restrict__ xp,
                                                          const __bf16* __restrict__ hp,
                                                          const float* __restrict__ h,
                                                          const __bf16* __restrict__ wc,
                                                          const float* __restrict__ b_ih,
                                                          const float* __restrict__ b_hh,
                                                          const __bf16* __restrict__ wp,
                                                          const float* __restrict__ b_proj,
                                                          float* __restrict__ out,
                                                          float* __restrict__ hout,
                                                          __bf16* __restrict__ hpack_out,
                                                          int t) {
    __shared__ __bf16 lds[64 * 66];
    const int lane = threadIdx.x & 31;
    const int wave = threadIdx.x >> 5;
    const int mi = wave & 3, ni = wave >> 2;

    if (blockIdx.y == 8) {
        // ---- projection path: out[:, :, t-1] = h_t @ w_proj + b_proj ----
        if (t == 0) return;
        const int mtile = blockIdx.x * 4 + mi;     // 0..31
        const int nt0 = ni, nt1 = ni + 4;          // two 16-col tiles each
        v8f acc0 = {}, acc1 = {};
#pragma unroll 4
        for (int kt = 0; kt < KT; ++kt) {
            v16bf a  = load_a_packed(hp, mtile, kt, lane);
            v16bf b0 = *(const v16bf*)(wp + (((size_t)nt0 * KT + kt) * 512) + lane * 16);
            v16bf b1 = *(const v16bf*)(wp + (((size_t)nt1 * KT + kt) * 512) + lane * 16);
            acc0 = __builtin_amdgcn_wmma_f32_16x16x32_bf16(false, a, false, b0, (short)0, acc0, false, false);
            acc1 = __builtin_amdgcn_wmma_f32_16x16x32_bf16(false, a, false, b1, (short)0, acc1, false, false);
        }
        const int rowOff = mtile * 16 + ((lane & 16) ? 8 : 0);
#pragma unroll 2
        for (int half = 0; half < 2; ++half) {
            const v8f& acc = half ? acc1 : acc0;
            const int v = (half ? nt1 : nt0) * 16 + (lane & 15);
            if (v < Vn) {
                const float bias = b_proj[v];
#pragma unroll
                for (int e = 0; e < 8; ++e)
                    out[((size_t)(rowOff + e) * Vn + v) * Tn + (t - 1)] = acc[e] + bias;
            }
        }
        return;
    }

    // ---- gate path ----
    const int mt = blockIdx.x * 4 + mi;            // A mtile (batch rows /16)
    const int ntg = blockIdx.y * 4 + ni;           // 16-col tile within a gate

    v8f acc[6] = {};
    if (xp == hp) {                                // t>=1: x_t == h_t
#pragma unroll 4
        for (int kt = 0; kt < KT; ++kt) {
            v16bf ah = load_a_packed(hp, mt, kt, lane);
#pragma unroll
            for (int g = 0; g < 6; ++g) {
                const __bf16* bp = wc + ((((size_t)g * 32 + ntg) * KT + kt) * 512) + lane * 16;
                v16bf bf = *(const v16bf*)bp;
                acc[g] = __builtin_amdgcn_wmma_f32_16x16x32_bf16(false, ah, false, bf, (short)0, acc[g], false, false);
            }
        }
    } else {
#pragma unroll 4
        for (int kt = 0; kt < KT; ++kt) {
            v16bf ax = load_a_packed(xp, mt, kt, lane);
            v16bf ah = load_a_packed(hp, mt, kt, lane);
#pragma unroll
            for (int g = 0; g < 6; ++g) {
                const __bf16* bp = wc + ((((size_t)g * 32 + ntg) * KT + kt) * 512) + lane * 16;
                v16bf bf = *(const v16bf*)bp;
                v16bf af = (g < 3) ? ax : ah;
                acc[g] = __builtin_amdgcn_wmma_f32_16x16x32_bf16(false, af, false, bf, (short)0, acc[g], false, false);
            }
        }
    }

    const int n = ntg * 16 + (lane & 15);
    const float bir = b_ih[n], biz = b_ih[Hn + n], bin = b_ih[2 * Hn + n];
    const float bhr = b_hh[n], bhz = b_hh[Hn + n], bhn = b_hh[2 * Hn + n];
    const int rowOff = blockIdx.x * 64 + mi * 16 + ((lane & 16) ? 8 : 0);
    const int rl0 = mi * 16 + ((lane & 16) ? 8 : 0);
    const int cl  = ni * 16 + (lane & 15);
#pragma unroll
    for (int e = 0; e < 8; ++e) {
        const int m = rowOff + e;
        const float hv = h[(size_t)m * Hn + n];
        const float r  = sigmoidf_(acc[0][e] + bir + acc[3][e] + bhr);
        const float z  = sigmoidf_(acc[1][e] + biz + acc[4][e] + bhz);
        const float nn = tanhf(acc[2][e] + bin + r * (acc[5][e] + bhn));
        const float hv2 = (1.0f - z) * nn + z * hv;
        hout[(size_t)m * Hn + n] = hv2;
        lds[(rl0 + e) * 66 + cl] = f2bf(hv2);
    }
    __syncthreads();
    tile_pack_store(lds, hpack_out, threadIdx.x, blockIdx.x, blockIdx.y);
}

// ---------------- tail projection (h_200 -> out[:, :, 199]) ----------------

__global__ __launch_bounds__(128, 1) void proj_kernel(const __bf16* __restrict__ hp,
                                                      const __bf16* __restrict__ wp,
                                                      const float* __restrict__ b_proj,
                                                      float* __restrict__ out, int t) {
    const int lane = threadIdx.x & 31;
    const int wave = threadIdx.x >> 5;             // 0..3
    const int mtile = blockIdx.x * 4 + wave;       // 0..31
    const int nt = blockIdx.y;                     // 0..6
    v8f acc = {};
#pragma unroll
    for (int kt = 0; kt < KT; ++kt) {
        v16bf a = load_a_packed(hp, mtile, kt, lane);
        v16bf b = *(const v16bf*)(wp + (((size_t)nt * KT + kt) * 512) + lane * 16);
        acc = __builtin_amdgcn_wmma_f32_16x16x32_bf16(false, a, false, b, (short)0, acc, false, false);
    }
    const int v = nt * 16 + (lane & 15);
    if (v < Vn) {
        const float bias = b_proj[v];
        const int rowOff = mtile * 16 + ((lane & 16) ? 8 : 0);
#pragma unroll
        for (int e = 0; e < 8; ++e)
            out[((size_t)(rowOff + e) * Vn + v) * Tn + t] = acc[e] + bias;
    }
}

// ---------------------------------------------------------------------------

extern "C" void kernel_launch(void* const* d_in, const int* in_sizes, int n_in,
                              void* d_out, int out_size, void* d_ws, size_t ws_size,
                              hipStream_t stream) {
    (void)in_sizes; (void)n_in; (void)out_size; (void)ws_size;
    const float* feat   = (const float*)d_in[0];
    const float* w_hp   = (const float*)d_in[1];
    const float* b_hp   = (const float*)d_in[2];
    const float* embed  = (const float*)d_in[3];
    const float* w_ih   = (const float*)d_in[4];
    const float* w_hh   = (const float*)d_in[5];
    const float* b_ih   = (const float*)d_in[6];
    const float* b_hh   = (const float*)d_in[7];
    const float* w_proj = (const float*)d_in[8];
    const float* b_proj = (const float*)d_in[9];
    float* out = (float*)d_out;

    // workspace layout (~8.6 MiB)
    char* ws = (char*)d_ws;
    float*  hA   = (float*) (ws + (size_t)0);                 // 1 MiB
    float*  hB   = (float*) (ws + ((size_t)1 << 20));         // 1 MiB
    __bf16* hpA  = (__bf16*)(ws + ((size_t)2 << 20));         // 512 KiB
    __bf16* hpB  = (__bf16*)(ws + ((size_t)2 << 20) + ((size_t)512 << 10));
    __bf16* x0p  = (__bf16*)(ws + ((size_t)3 << 20));         // 512 KiB
    __bf16* whp  = (__bf16*)(ws + ((size_t)7 << 19));         // 2 MiB @ 3.5 MiB
    __bf16* wc   = (__bf16*)(ws + ((size_t)11 << 19));        // 3 MiB @ 5.5 MiB
    __bf16* wp   = (__bf16*)(ws + ((size_t)17 << 19));        // 128 KiB @ 8.5 MiB

    // pack weights + init state
    pack_whp_kernel  <<<(32 * 64 * 512) / 256, 256, 0, stream>>>(w_hp, whp);
    pack_wc_kernel   <<<(6 * 32 * 16 * 512) / 256, 256, 0, stream>>>(w_ih, w_hh, wc);
    pack_wproj_kernel<<<(8 * 16 * 512) / 256, 256, 0, stream>>>(w_proj, wp);
    pack_x0_kernel   <<<(32 * 16 * 512) / 256, 256, 0, stream>>>(embed, x0p);
    h0_kernel        <<<dim3(8, 8), 512, 0, stream>>>(feat, whp, b_hp, hA, hpA);

    // 200-step recurrence; stream ordering provides the grid-wide sync.
    // Step t also projects its input h_t into out[:, :, t-1] (t>=1).
    float*  hcur = hA;   float*  hnext = hB;
    __bf16* pcur = hpA;  __bf16* pnext = hpB;
    for (int t = 0; t < Tn; ++t) {
        const __bf16* xin = (t == 0) ? x0p : pcur;
        gru_step_kernel<<<dim3(8, 9), 512, 0, stream>>>(xin, pcur, hcur, wc, b_ih, b_hh,
                                                        wp, b_proj, out, hnext, pnext, t);
        float*  t0 = hcur; hcur = hnext; hnext = t0;
        __bf16* t1 = pcur; pcur = pnext; pnext = t1;
    }
    // h_200 -> out[:, :, 199]
    proj_kernel<<<dim3(8, 7), 128, 0, stream>>>(pcur, wp, b_proj, out, Tn - 1);
}
